// VDP_Full_FullyConnected_26499948216842
// MI455X (gfx1250) — matrix operand order repaired and checked by reading, not compiled, not run on hardware
//
#include <hip/hip_runtime.h>
#include <hip/hip_bf16.h>

// ---------------------------------------------------------------------------
// VDP Full FullyConnected on MI455X (gfx1250, wave32, WMMA)
//
// Heavy part: sigma_out[b] = W (384x512) @ sigma[b] (512x512) @ W^T  (135 GFLOP)
// Memory floor: read sigma 402MB + write out 226MB ~= 27us @ 23.3 TB/s.
// f32 VALU compute would be >1ms -> use v_wmma_f32_16x16x32_f16 (f16 in,
// f32 accumulate). Both GEMMs fused per batch (T stays in LDS).
// This round: B operands staged fragment-major in LDS so every WMMA operand
// is 2x ds_load_b128 (no scalar u16 gather / v_mov_b16 packing).
// ---------------------------------------------------------------------------

#define B_    384
#define IN_   512
#define OUT_  384

typedef __attribute__((ext_vector_type(16))) _Float16 v16h;
typedef __attribute__((ext_vector_type(8)))  _Float16 v8h;
typedef __attribute__((ext_vector_type(8)))  float    v8f;

// Row strides (halves) for row-major LDS arrays (A operands), 16B-multiple
#define LW_STR  520   // W block 64 x 512
#define LT_STR  72    // T tile  64 x 64

// Fragment-major 32x16 B tile: 64 16B-units + pad every 8 units -> 72 units
#define BT_TILE 576   // halves per padded B tile (72 * 8)

// ---- WMMA fragment helpers (CDNA5 ISA 7.12.2 lane layouts) ----------------

// A (16x32 f16), row-major LDS: lane holds M=lane&15; two contiguous 8-half
// K-runs at k0 + 8*(lane>>4) and +16.  Both runs 16B aligned.
__device__ __forceinline__ v16h load_frag_a(const _Float16* base, int stride,
                                            int row0, int k0, int lane) {
  const _Float16* p = base + (row0 + (lane & 15)) * stride + k0 + ((lane >> 4) << 3);
  const v8h lo = *(const v8h*)p;
  const v8h hi = *(const v8h*)(p + 16);
  return __builtin_shufflevector(lo, hi, 0, 1, 2, 3, 4, 5, 6, 7,
                                 8, 9, 10, 11, 12, 13, 14, 15);
}

// Position of element (kl in 0..31, nl in 0..15) inside a padded B tile.
// B lane layout: lane' = nl + 16*bit3(kl), element h = 8*bit4(kl) + (kl&7).
// 16B-unit index u = lane'*2 + bit4(kl); pad: phys unit = u + (u>>3).
__device__ __forceinline__ int bfrag_off(int nl, int kl) {
  const int lanep = nl + (((kl >> 3) & 1) << 4);
  const int u = lanep * 2 + (kl >> 4);
  return (u + (u >> 3)) * 8 + (kl & 7);
}

// B (32x16 f16) from a fragment-major tile: two contiguous 16B loads per lane.
__device__ __forceinline__ v16h load_frag_b(const _Float16* tile, int lane) {
  const int u = lane * 2;                       // even -> (u>>3) == ((u+1)>>3)
  const _Float16* p = tile + (u + (u >> 3)) * 8;
  const v8h lo = *(const v8h*)p;
  const v8h hi = *(const v8h*)(p + 8);
  return __builtin_shufflevector(lo, hi, 0, 1, 2, 3, 4, 5, 6, 7,
                                 8, 9, 10, 11, 12, 13, 14, 15);
}

__device__ __forceinline__ v8f wmma_f16(v16h a, v16h b, v8f c) {
  return __builtin_amdgcn_wmma_f32_16x16x32_f16(false, a, false, b,
                                                (short)0, c, false, false);
}

// ---------------------------------------------------------------------------
// prep: d[b,i] = sigma[b,i,i] + mu[b,i]^2 ; sp = softplus(sigma_weight);
//       W -> f16 (row major and transposed) for the WMMA kernel.
// ---------------------------------------------------------------------------
__global__ __launch_bounds__(256) void prep_kernel(
    const float* __restrict__ mu, const float* __restrict__ sigma,
    const float* __restrict__ W,  const float* __restrict__ sw,
    float* __restrict__ ws_d, float* __restrict__ ws_sp,
    _Float16* __restrict__ Wh, _Float16* __restrict__ Wt) {
  const int idx = blockIdx.x * blockDim.x + threadIdx.x;
  if (idx >= B_ * IN_) return;
  const int r = idx >> 9;      // b (also o: B_ == OUT_)
  const int i = idx & (IN_ - 1);
  const float m = mu[idx];
  ws_d[idx]  = sigma[((size_t)r * IN_ + i) * IN_ + i] + m * m;
  ws_sp[idx] = log1pf(expf(sw[idx]));      // softplus, sw in [-12,-2.2]: safe
  const float w = W[idx];
  Wh[idx]                  = (_Float16)w;            // Wh[o][i]
  Wt[(size_t)i * OUT_ + r] = (_Float16)w;            // Wt[i][o] = W[o][i]
}

// ---------------------------------------------------------------------------
// small_gemms: mu_f[b,o] = mu[b,:].W[o,:] + bias[o]
//              s13[b,o]  = d[b,:].sp[o,:] + (b==o)*sigma_bias[o]
// 300 MFLOP total -> scalar FMA fine. block = 384 threads (12 waves), b/block.
// ---------------------------------------------------------------------------
__global__ __launch_bounds__(384) void small_gemms_kernel(
    const float* __restrict__ mu, const float* __restrict__ W,
    const float* __restrict__ bias, const float* __restrict__ sbias,
    const float* __restrict__ ws_d, const float* __restrict__ ws_sp,
    float* __restrict__ out_mu, float* __restrict__ ws_s13) {
  const int b = blockIdx.x;
  const int o = threadIdx.x;
  const float* __restrict__ murow = mu    + (size_t)b * IN_;
  const float* __restrict__ drow  = ws_d  + (size_t)b * IN_;
  const float* __restrict__ wrow  = W     + (size_t)o * IN_;
  const float* __restrict__ sprow = ws_sp + (size_t)o * IN_;
  float a1 = 0.f, a2 = 0.f;
#pragma unroll 4
  for (int i = 0; i < IN_; ++i) {
    a1 = fmaf(murow[i], wrow[i], a1);
    a2 = fmaf(drow[i], sprow[i], a2);
  }
  out_mu[b * OUT_ + o] = a1 + bias[o];
  ws_s13[b * OUT_ + o] = a2 + (b == o ? sbias[o] : 0.f);
}

// ---------------------------------------------------------------------------
// heavy: out[b, ob*64 : +64, 0:384] = W_blk @ sigma[b] @ W^T (+ s13 diag)
// 8 waves / block; j in 64-wide chunks:
//   GEMM1 (WMMA): T(64x64) = W_blk(64x512) @ sigma[:, j:j+64]   -> LDS (f16)
//   GEMM2 (WMMA): acc(64x384) += T @ W^T[j:j+64, :]             -> VGPRs (f32)
// GEMM1: wave owns 2 N-tiles of one M row (A fragment shared).
// GEMM2: wave (wo,wp) owns o in [wo*32,+32), p in [wp*96,+96): 2x6 tiles.
// ---------------------------------------------------------------------------
__global__ __launch_bounds__(256) void heavy_kernel(
    const float* __restrict__ sigma, const _Float16* __restrict__ Wh,
    const _Float16* __restrict__ Wt, const float* __restrict__ s13,
    float* __restrict__ out) {
  extern __shared__ __align__(16) _Float16 smem[];
  _Float16* sW  = smem;                      // 64 x LW_STR       (row major)
  _Float16* sS  = sW  + 64 * LW_STR;         // 16x4 B tiles      (frag major)
  _Float16* sWt = sS  + 64 * BT_TILE;        // 2x24 B tiles      (frag major)
  _Float16* sT  = sWt + 48 * BT_TILE;        // 64 x LT_STR       (row major)

  const int ob   = blockIdx.x;               // 0..5   (o block)
  const int b    = blockIdx.y;               // 0..383 (batch); adjacent blocks share sigma_b in L2
  const int tid  = threadIdx.x;
  const int lane = tid & 31;
  const int w    = tid >> 5;                 // wave 0..7
  const int wo   = w >> 2;                   // 0..1  (GEMM2 o half)
  const int wp   = w & 3;                    // 0..3  (GEMM2 p quarter)
  const int tm   = w >> 1;                   // 0..3  (GEMM1 M tile)
  const int tn0  = (w & 1) << 1;             // 0 / 2 (GEMM1 first N tile)

  // Load the 64x512 f16 W block once.
  for (int idx = tid; idx < 64 * IN_; idx += 256) {
    const int r = idx >> 9, k = idx & (IN_ - 1);
    sW[r * LW_STR + k] = Wh[(size_t)(ob * 64 + r) * IN_ + k];
  }

  v8f acc[2][6] = {};

  const float* __restrict__ sigb = sigma + (size_t)b * IN_ * IN_;

  const int mlo = (lane >> 4) << 3;          // C/D row offset for this lane
  const int ncd = lane & 15;                 // C/D column for this lane

  for (int j0 = 0; j0 < IN_; j0 += 64) {
    __syncthreads();   // prior iteration consumed sS/sWt/sT (also fences sW)

    // stage sigma[:, j0:j0+64] -> f16, fragment-major 32x16 tiles
    for (int idx = tid; idx < IN_ * 64; idx += 256) {
      const int i = idx >> 6, jj = idx & 63;           // contiguous jj reads
      const float v = sigb[(size_t)i * IN_ + j0 + jj];
      const int tile = ((i >> 5) << 2) + (jj >> 4);    // kt*4 + nt
      sS[tile * BT_TILE + bfrag_off(jj & 15, i & 31)] = (_Float16)v;
    }
    // stage W^T[j0:j0+64, 0:384] -> fragment-major 32x16 tiles
    for (int idx = tid; idx < 64 * OUT_; idx += 256) {
      const int jj = idx / OUT_, p = idx % OUT_;       // contiguous p reads
      const int tile = (jj >> 5) * 24 + (p >> 4);      // kt*24 + nt
      sWt[tile * BT_TILE + bfrag_off(p & 15, jj & 31)] =
          Wt[(size_t)(j0 + jj) * OUT_ + p];
    }
    // prefetch next sigma chunk while this one is consumed
    if (j0 + 64 < IN_) {
      for (int r = tid; r < IN_; r += 256)
        __builtin_prefetch(&sigb[(size_t)r * IN_ + j0 + 64], 0, 0);
    }
    __syncthreads();

    // GEMM1: wave computes T tiles (tm, tn0) and (tm, tn0+1); K = 512
    {
      v8f c0 = {}, c1 = {};
      for (int ks = 0; ks < IN_; ks += 32) {
        const int kt4 = (ks >> 5) << 2;
        v16h a  = load_frag_a(sW, LW_STR, tm * 16, ks, lane);
        v16h b0 = load_frag_b(sS + (kt4 + tn0) * BT_TILE, lane);
        v16h b1 = load_frag_b(sS + (kt4 + tn0 + 1) * BT_TILE, lane);
        c0 = wmma_f16(a, b0, c0);
        c1 = wmma_f16(a, b1, c1);
      }
#pragma unroll
      for (int r = 0; r < 8; ++r) {
        const int row = (tm * 16 + mlo + r) * LT_STR;
        sT[row + tn0 * 16 + ncd]       = (_Float16)c0[r];
        sT[row + (tn0 + 1) * 16 + ncd] = (_Float16)c1[r];
      }
    }
    __syncthreads();

    // GEMM2: acc(2x6 tiles) += T(64x64) @ sWt(64x384), K = 64
#pragma unroll
    for (int ks = 0; ks < 64; ks += 32) {
      v16h a0 = load_frag_a(sT, LT_STR, wo * 32,      ks, lane);
      v16h a1 = load_frag_a(sT, LT_STR, wo * 32 + 16, ks, lane);
#pragma unroll
      for (int tj = 0; tj < 6; ++tj) {
        const int nt = wp * 6 + tj;
        v16h bb = load_frag_b(sWt + ((ks >> 5) * 24 + nt) * BT_TILE, lane);
        acc[0][tj] = wmma_f16(a0, bb, acc[0][tj]);
        acc[1][tj] = wmma_f16(a1, bb, acc[1][tj]);
      }
    }
  }

  // Epilogue: add s13 on the diagonal, store f32
#pragma unroll
  for (int ti = 0; ti < 2; ++ti)
#pragma unroll
    for (int tj = 0; tj < 6; ++tj)
#pragma unroll
      for (int r = 0; r < 8; ++r) {
        const int o = ob * 64 + wo * 32 + ti * 16 + mlo + r;
        const int p = wp * 96 + tj * 16 + ncd;
        float v = acc[ti][tj][r];
        if (o == p) v += s13[b * OUT_ + o];
        out[((size_t)b * OUT_ + o) * OUT_ + p] = v;
      }
}

// ---------------------------------------------------------------------------
extern "C" void kernel_launch(void* const* d_in, const int* in_sizes, int n_in,
                              void* d_out, int out_size, void* d_ws, size_t ws_size,
                              hipStream_t stream) {
  const float* mu    = (const float*)d_in[0];
  const float* sigma = (const float*)d_in[1];
  const float* W     = (const float*)d_in[2];
  const float* bias  = (const float*)d_in[3];
  const float* sw    = (const float*)d_in[4];
  const float* sbias = (const float*)d_in[5];
  float* out = (float*)d_out;

  // workspace layout (floats): d[B,IN] | sp[OUT,IN] | s13[B,OUT] | Wh f16 | Wt f16
  float* wsf       = (float*)d_ws;
  float* ws_d      = wsf;
  float* ws_sp     = wsf + (size_t)B_ * IN_;
  float* ws_s13    = wsf + 2 * (size_t)B_ * IN_;
  _Float16* ws_Wh  = (_Float16*)(ws_s13 + (size_t)B_ * OUT_);
  _Float16* ws_Wt  = ws_Wh + (size_t)OUT_ * IN_;

  prep_kernel<<<dim3((B_ * IN_ + 255) / 256), 256, 0, stream>>>(
      mu, sigma, W, sw, ws_d, ws_sp, ws_Wh, ws_Wt);

  small_gemms_kernel<<<dim3(B_), OUT_, 0, stream>>>(
      mu, W, bias, sbias, ws_d, ws_sp, out, ws_s13);

  const size_t smem_bytes =
      (size_t)(64 * LW_STR + 64 * BT_TILE + 48 * BT_TILE + 64 * LT_STR) *
      sizeof(_Float16);  // 204800 B of the 320 KB WGP LDS
  heavy_kernel<<<dim3(OUT_ / 64, B_), 256, smem_bytes, stream>>>(
      sigma, ws_Wh, ws_Wt, ws_s13, out + (size_t)B_ * OUT_);
}
